// SwinTransformerBlock3D_54339926229587
// MI455X (gfx1250) — matrix operand order repaired and checked by reading, not compiled, not run on hardware
//
#include <hip/hip_runtime.h>
#include <hip/hip_bf16.h>
#include <math.h>

// ---------------- problem constants ----------------
#define Bb   1
#define Dd   16
#define Hh   56
#define Www  56
#define Cc   256
#define WDw  2
#define WHw  7
#define WWw  7
#define Nn_  98          // tokens per window
#define NHh  8
#define NPp  4
#define HDd  32
#define NWIN 512
#define Mtok 50176       // B*D*H*W = NWIN*N

typedef __attribute__((ext_vector_type(16))) __bf16 bf16x16;
typedef __attribute__((ext_vector_type(8)))  __bf16 bf16x8;
typedef __attribute__((ext_vector_type(8)))  float  floatx8;

// ---------------- workspace layout (bytes) ----------------
//  weights are stored TRANSPOSED [N][K] in bf16 so GEMM B-tiles are k-contiguous
//  XN  : f32  M*C   xn (LN1 out)            -> reused for z (proj out)
//  V   : f32  M*C   value                   -> reused for y (attn residual out)
//  Q   : bf16 M*C   q
//  BB  : bf16 M*C   xn_bf16 -> attnpre -> h
//  OA  : f32  M*128 off|aw  -> t (bf16 M*C, same bytes)
//  G   : bf16 M*1024 gelu(fc1) out
static constexpr size_t OFF_WVAL  = 0;                       // 256*256*2
static constexpr size_t OFF_WOA   = OFF_WVAL  + 131072;      // 128*256*2
static constexpr size_t OFF_WOUT  = OFF_WOA   + 65536;
static constexpr size_t OFF_WPROJ = OFF_WOUT  + 131072;
static constexpr size_t OFF_WFC1  = OFF_WPROJ + 131072;      // 1024*256*2
static constexpr size_t OFF_WFC2  = OFF_WFC1  + 524288;      // 256*1024*2
static constexpr size_t OFF_BOA   = OFF_WFC2  + 524288;      // 128*4 (padded)
static constexpr size_t OFF_DIAG  = OFF_BOA   + 1024;        // 50176*4
static constexpr size_t OFF_XN    = OFF_DIAG  + 200704;      // f32 M*C
static constexpr size_t OFF_V     = OFF_XN    + 51380224;    // f32 M*C
static constexpr size_t OFF_Q     = OFF_V     + 51380224;    // bf16 M*C
static constexpr size_t OFF_BB    = OFF_Q     + 25690112;    // bf16 M*C
static constexpr size_t OFF_OA    = OFF_BB    + 25690112;    // f32 M*128 / bf16 M*C
static constexpr size_t OFF_G     = OFF_OA    + 25690112;    // bf16 M*1024

// ---------------- helpers ----------------
__device__ __forceinline__ __bf16 f2bf(float f) {
  unsigned u = __builtin_bit_cast(unsigned, f);
  unsigned r = (u + 0x7FFFu + ((u >> 16) & 1u)) >> 16;
  unsigned short s = (unsigned short)r;
  return __builtin_bit_cast(__bf16, s);
}
struct BF16X8X2 { bf16x8 lo; bf16x8 hi; };
__device__ __forceinline__ bf16x16 make16(bf16x8 lo, bf16x8 hi) {
  BF16X8X2 t{lo, hi};
  return __builtin_bit_cast(bf16x16, t);
}
__device__ __forceinline__ void store_out(float* p, float v)  { *p = v; }
__device__ __forceinline__ void store_out(__bf16* p, float v) { *p = f2bf(v); }
__device__ __forceinline__ float gelu_tanh(float x) {
  float x3 = x * x * x;
  return 0.5f * x * (1.f + tanhf(0.79788456080286535588f * (x + 0.044715f * x3)));
}

// ---------------- weight packing (f32 [K][N] -> bf16 [N][K]) ----------------
__global__ __launch_bounds__(256) void pack_bf16_t(const float* __restrict__ s,
                                                   __bf16* __restrict__ d, int K, int N) {
  int i = blockIdx.x * 256 + threadIdx.x;
  if (i < K * N) {
    int k = i / N, n = i % N;
    d[(size_t)n * K + k] = f2bf(s[i]);
  }
}

__global__ __launch_bounds__(256) void pack_cat_t(const float* __restrict__ offw,
                                                  const float* __restrict__ aww,
                                                  const float* __restrict__ offb,
                                                  const float* __restrict__ awb,
                                                  __bf16* __restrict__ wd,
                                                  float* __restrict__ bd) {
  int i = blockIdx.x * 256 + threadIdx.x;      // 256*128
  if (i >= 256 * 128) return;
  int k = i >> 7, j = i & 127;
  float v = (j < 96) ? offw[k * 96 + j] : aww[k * 32 + (j - 96)];
  wd[(size_t)j * 256 + k] = f2bf(v);           // transposed [128][256]
  if (k == 0) bd[j] = (j < 96) ? offb[j] : awb[j - 96];
}

// ---------------- position-embedding diagonal ----------------
__global__ __launch_bounds__(256) void diag_kernel(const float* __restrict__ mask,
                                                   float* __restrict__ diag) {
  int t = blockIdx.x * 256 + threadIdx.x;      // NWIN*N = 50176 exactly
  const float* mp = mask + (size_t)t * Nn_;
  int n = t % Nn_;
  int zd = n / 49, yd = (n / 7) % 7, xd = n % 7;
  float mval = mp[zd * 49 + yd * 7 + xd];
  float csz = 0.f, csy = 0.f, csx = 0.f;
  for (int k = 0; k <= zd; ++k) csz += mp[k * 49 + yd * 7 + xd];
  for (int k = 0; k <= yd; ++k) csy += mp[zd * 49 + k * 7 + xd];
  for (int k = 0; k <= xd; ++k) csx += mp[zd * 49 + yd * 7 + k];
  const float PI = 3.14159265358979323846f;
  float ze = csz * mval / PI;
  float ye = csy * mval / PI;
  float xe = csx * mval / PI;
  diag[t] = mval * (sinf(xe) + sinf(PI * ye) + ze);
}

// ---------------- LN1 + window partition (wave per token) ----------------
__global__ __launch_bounds__(256) void ln1_kernel(const float* __restrict__ x,
                                                  const float* __restrict__ g,
                                                  const float* __restrict__ b,
                                                  const float* __restrict__ diag,
                                                  float* __restrict__ xn,
                                                  __bf16* __restrict__ xnb,
                                                  __bf16* __restrict__ qb) {
  int m = (blockIdx.x << 3) + (threadIdx.x >> 5);   // window-order token
  int lane = threadIdx.x & 31;
  int win = m / Nn_, n = m % Nn_;
  int wd = win >> 6, wh = (win >> 3) & 7, wwi = win & 7;
  int id = n / 49, ih = (n / 7) % 7, iw = n % 7;
  int d = wd * WDw + id, h = wh * WHw + ih, w = wwi * WWw + iw;
  const float* src = x + (((size_t)d * Hh + h) * Www + w) * Cc + lane * 8;
  float v[8];
  float4 a0 = *(const float4*)(src);
  float4 a1 = *(const float4*)(src + 4);
  v[0]=a0.x; v[1]=a0.y; v[2]=a0.z; v[3]=a0.w;
  v[4]=a1.x; v[5]=a1.y; v[6]=a1.z; v[7]=a1.w;
  float s = 0.f, s2 = 0.f;
#pragma unroll
  for (int i = 0; i < 8; ++i) { s += v[i]; s2 += v[i] * v[i]; }
#pragma unroll
  for (int off = 16; off; off >>= 1) { s += __shfl_xor(s, off, 32); s2 += __shfl_xor(s2, off, 32); }
  float mean = s * (1.f / 256.f);
  float rs = rsqrtf(s2 * (1.f / 256.f) - mean * mean + 1e-5f);
  float dg = diag[m];
  size_t o = (size_t)m * Cc + lane * 8;
#pragma unroll
  for (int i = 0; i < 8; ++i) {
    int c = lane * 8 + i;
    float xv = (v[i] - mean) * rs * g[c] + b[c];
    xn[o + i] = xv;
    xnb[o + i] = f2bf(xv);
    qb[o + i] = f2bf(xv + dg);
  }
}

// ---------------- bf16 WMMA GEMM: C = A*Bt^T + bias (+resid) (+gelu) ----------------
// A: [M][K] bf16, Bt: [N][K] bf16 (weights pre-transposed) -> both tiles k-contiguous.
// 128x128 block tile, 8 waves, each wave 32x64 = 2x4 WMMA 16x16x32 tiles.
// LDS double buffered, one barrier per K step.
template <int EPI, int HAS_RES, typename OutT>
__global__ __launch_bounds__(256) void gemm_wmma(const __bf16* __restrict__ A,
                                                 const __bf16* __restrict__ Bt,
                                                 const float* __restrict__ bias,
                                                 const float* __restrict__ resid,
                                                 OutT* __restrict__ Cout,
                                                 int M, int N, int K) {
  constexpr int SA = 40;    // padded row stride (elems): 80B rows, 16B aligned, conflict-free
  __shared__ __attribute__((aligned(16))) __bf16 lA[2][128 * SA];
  __shared__ __attribute__((aligned(16))) __bf16 lB[2][128 * SA];
  const int tid = threadIdx.x;
  const int lane = tid & 31, wave = tid >> 5;
  const int wm = wave & 3, wn = wave >> 2;
  const int half = lane >> 4, l15 = lane & 15;
  const int m0 = blockIdx.y * 128, n0 = blockIdx.x * 128;
  const int r2 = tid >> 1;            // tile row 0..127 (two 16B segs per thread)
  const int s2 = (tid & 1) * 2;       // seg 0 or 2

  uint4 a0, a1, b0, b1;
  const size_t gaBase = (size_t)(m0 + r2) * K + s2 * 8;
  const size_t gbBase = (size_t)(n0 + r2) * K + s2 * 8;

  floatx8 acc[2][4] = {};
  const int nsteps = K >> 5;

  {  // prologue: tile 0
    const uint4* pa = reinterpret_cast<const uint4*>(A + gaBase);
    a0 = pa[0]; a1 = pa[1];
    const uint4* pb = reinterpret_cast<const uint4*>(Bt + gbBase);
    b0 = pb[0]; b1 = pb[1];
    *reinterpret_cast<uint4*>(&lA[0][r2 * SA + s2 * 8]) = a0;
    *reinterpret_cast<uint4*>(&lA[0][r2 * SA + (s2 + 1) * 8]) = a1;
    *reinterpret_cast<uint4*>(&lB[0][r2 * SA + s2 * 8]) = b0;
    *reinterpret_cast<uint4*>(&lB[0][r2 * SA + (s2 + 1) * 8]) = b1;
  }
  __syncthreads();

  for (int s = 0; s < nsteps; ++s) {
    const int cur = s & 1;
    if (s + 1 < nsteps) {              // issue next-tile global loads early
      const int kk = (s + 1) << 5;
      const uint4* pa = reinterpret_cast<const uint4*>(A + gaBase + kk);
      a0 = pa[0]; a1 = pa[1];
      const uint4* pb = reinterpret_cast<const uint4*>(Bt + gbBase + kk);
      b0 = pb[0]; b1 = pb[1];
    }

    bf16x16 af[2], bfr[4];
#pragma unroll
    for (int t = 0; t < 2; ++t) {      // A frag: k = [half*8..+7] and [16+half*8..+7]
      const __bf16* base = &lA[cur][(wm * 32 + t * 16 + l15) * SA + half * 8];
      af[t] = make16(*reinterpret_cast<const bf16x8*>(base),
                     *reinterpret_cast<const bf16x8*>(base + 16));
    }
#pragma unroll
    for (int j = 0; j < 4; ++j) {      // B frag: k = [half*8..+7] and [16+half*8..+7]
      const __bf16* base = &lB[cur][(wn * 64 + j * 16 + l15) * SA + half * 8];
      bfr[j] = make16(*reinterpret_cast<const bf16x8*>(base),
                      *reinterpret_cast<const bf16x8*>(base + 16));
    }
#pragma unroll
    for (int t = 0; t < 2; ++t)
#pragma unroll
      for (int j = 0; j < 4; ++j)
        acc[t][j] = __builtin_amdgcn_wmma_f32_16x16x32_bf16(
            false, af[t], false, bfr[j], (short)0, acc[t][j], false, false);

    if (s + 1 < nsteps) {              // stage next tile into the idle buffer
      const int nxt = cur ^ 1;
      *reinterpret_cast<uint4*>(&lA[nxt][r2 * SA + s2 * 8]) = a0;
      *reinterpret_cast<uint4*>(&lA[nxt][r2 * SA + (s2 + 1) * 8]) = a1;
      *reinterpret_cast<uint4*>(&lB[nxt][r2 * SA + s2 * 8]) = b0;
      *reinterpret_cast<uint4*>(&lB[nxt][r2 * SA + (s2 + 1) * 8]) = b1;
      __syncthreads();
    }
  }

  // epilogue: hoisted base pointers, row stepping by +N
#pragma unroll
  for (int t = 0; t < 2; ++t) {
    const int row0 = m0 + wm * 32 + t * 16 + half * 8;
#pragma unroll
    for (int j = 0; j < 4; ++j) {
      const int col = n0 + wn * 64 + j * 16 + l15;
      const float bv = bias[col];
      OutT* cp = Cout + (size_t)row0 * N + col;
      const float* rp = resid + (size_t)row0 * N + col;
#pragma unroll
      for (int r = 0; r < 8; ++r) {
        float v = acc[t][j][r] + bv;
        if (HAS_RES) v += rp[(size_t)r * N];
        if (EPI == 1) v = gelu_tanh(v);
        store_out(cp + (size_t)r * N, v);
      }
    }
  }
}

// ---------------- softmax + trilinear sampling (wave per token*head, lane=hd) ----------------
__global__ __launch_bounds__(256) void sample_kernel(const float* __restrict__ value,
                                                     const float* __restrict__ offaw,
                                                     __bf16* __restrict__ attnpre) {
  int wid = (blockIdx.x << 3) + (threadIdx.x >> 5);
  int lane = threadIdx.x & 31;
  int m = wid >> 3;         // token (window order)
  int head = wid & 7;
  int win = m / Nn_, n = m % Nn_;
  int id = n / 49, ih = (n / 7) % 7, iw = n % 7;
  float refx = (iw + 0.5f) / 7.f;
  float refy = (ih + 0.5f) / 7.f;
  float refz = (id + 0.5f) / 2.f;
  const float* oa = offaw + (size_t)m * 128;
  float l0 = oa[96 + head * 4 + 0], l1 = oa[96 + head * 4 + 1];
  float l2 = oa[96 + head * 4 + 2], l3 = oa[96 + head * 4 + 3];
  float lm = fmaxf(fmaxf(l0, l1), fmaxf(l2, l3));
  float e0 = expf(l0 - lm), e1 = expf(l1 - lm), e2 = expf(l2 - lm), e3 = expf(l3 - lm);
  float inv = 1.f / (e0 + e1 + e2 + e3);
  float aws[4] = {e0 * inv, e1 * inv, e2 * inv, e3 * inv};
  const float* vb = value + (size_t)win * Nn_ * Cc + head * HDd + lane;
  float acc = 0.f;
#pragma unroll
  for (int p = 0; p < 4; ++p) {
    float ox = oa[(head * 4 + p) * 3 + 0];
    float oy = oa[(head * 4 + p) * 3 + 1];
    float oz = oa[(head * 4 + p) * 3 + 2];
    float px = (refx + ox / 7.f) * 7.f - 0.5f;
    float py = (refy + oy / 7.f) * 7.f - 0.5f;
    float pz = (refz + oz / 2.f) * 2.f - 0.5f;
    float x0 = floorf(px), y0 = floorf(py), z0 = floorf(pz);
    float ap = aws[p];
#pragma unroll
    for (int dz = 0; dz < 2; ++dz)
#pragma unroll
      for (int dy = 0; dy < 2; ++dy)
#pragma unroll
        for (int dx = 0; dx < 2; ++dx) {
          float xi = x0 + dx, yi = y0 + dy, zi = z0 + dz;
          float wgt = (1.f - fabsf(px - xi)) * (1.f - fabsf(py - yi)) * (1.f - fabsf(pz - zi));
          if (xi >= 0.f && xi < 7.f && yi >= 0.f && yi < 7.f && zi >= 0.f && zi < 2.f) {
            int idx = (int)zi * 49 + (int)yi * 7 + (int)xi;
            acc += vb[(size_t)idx * Cc] * (wgt * ap);
          }
        }
  }
  attnpre[(size_t)m * Cc + head * HDd + lane] = f2bf(acc);
}

// ---------------- plain LN (wave per token) -> bf16 ----------------
__global__ __launch_bounds__(256) void ln_simple(const float* __restrict__ y,
                                                 const float* __restrict__ g,
                                                 const float* __restrict__ b,
                                                 __bf16* __restrict__ t) {
  int m = (blockIdx.x << 3) + (threadIdx.x >> 5);
  int lane = threadIdx.x & 31;
  const float* src = y + (size_t)m * Cc + lane * 8;
  float v[8];
  float4 a0 = *(const float4*)(src);
  float4 a1 = *(const float4*)(src + 4);
  v[0]=a0.x; v[1]=a0.y; v[2]=a0.z; v[3]=a0.w;
  v[4]=a1.x; v[5]=a1.y; v[6]=a1.z; v[7]=a1.w;
  float s = 0.f, s2 = 0.f;
#pragma unroll
  for (int i = 0; i < 8; ++i) { s += v[i]; s2 += v[i] * v[i]; }
#pragma unroll
  for (int off = 16; off; off >>= 1) { s += __shfl_xor(s, off, 32); s2 += __shfl_xor(s2, off, 32); }
  float mean = s * (1.f / 256.f);
  float rs = rsqrtf(s2 * (1.f / 256.f) - mean * mean + 1e-5f);
  size_t o = (size_t)m * Cc + lane * 8;
#pragma unroll
  for (int i = 0; i < 8; ++i) {
    int c = lane * 8 + i;
    t[o + i] = f2bf((v[i] - mean) * rs * g[c] + b[c]);
  }
}

// ---------------- window reverse + shortcut + LN2 ----------------
__global__ __launch_bounds__(256) void reverse_ln2(const float* __restrict__ xin,
                                                   const float* __restrict__ z,
                                                   const float* __restrict__ g,
                                                   const float* __restrict__ b,
                                                   float* __restrict__ outx,
                                                   __bf16* __restrict__ hbf) {
  int m = (blockIdx.x << 3) + (threadIdx.x >> 5);   // natural-order token
  int lane = threadIdx.x & 31;
  int d = m / (Hh * Www), h = (m / Www) % Hh, w = m % Www;
  int wd = d >> 1, id = d & 1;
  int wh = h / 7, ih = h % 7;
  int wwi = w / 7, iw = w % 7;
  int win = (wd * 8 + wh) * 8 + wwi;
  int n = (id * 7 + ih) * 7 + iw;
  size_t mw = ((size_t)win * Nn_ + n) * Cc + lane * 8;
  size_t mn = (size_t)m * Cc + lane * 8;
  float v[8];
#pragma unroll
  for (int i = 0; i < 8; ++i) v[i] = xin[mn + i] + z[mw + i];
#pragma unroll
  for (int i = 0; i < 8; ++i) outx[mn + i] = v[i];
  float s = 0.f, s2 = 0.f;
#pragma unroll
  for (int i = 0; i < 8; ++i) { s += v[i]; s2 += v[i] * v[i]; }
#pragma unroll
  for (int off = 16; off; off >>= 1) { s += __shfl_xor(s, off, 32); s2 += __shfl_xor(s2, off, 32); }
  float mean = s * (1.f / 256.f);
  float rs = rsqrtf(s2 * (1.f / 256.f) - mean * mean + 1e-5f);
#pragma unroll
  for (int i = 0; i < 8; ++i) {
    int c = lane * 8 + i;
    hbf[mn + i] = f2bf((v[i] - mean) * rs * g[c] + b[c]);
  }
}

// ---------------- launch ----------------
extern "C" void kernel_launch(void* const* d_in, const int* in_sizes, int n_in,
                              void* d_out, int out_size, void* d_ws, size_t ws_size,
                              hipStream_t stream) {
  const float* x      = (const float*)d_in[0];
  const float* mask   = (const float*)d_in[1];
  const float* n1g    = (const float*)d_in[2];
  const float* n1b    = (const float*)d_in[3];
  const float* val_w  = (const float*)d_in[4];
  const float* val_b  = (const float*)d_in[5];
  const float* off_w  = (const float*)d_in[6];
  const float* off_b  = (const float*)d_in[7];
  const float* aw_w   = (const float*)d_in[8];
  const float* aw_b   = (const float*)d_in[9];
  const float* out_w  = (const float*)d_in[10];
  const float* out_b  = (const float*)d_in[11];
  const float* ang    = (const float*)d_in[12];
  const float* anb    = (const float*)d_in[13];
  const float* proj_w = (const float*)d_in[14];
  const float* proj_b = (const float*)d_in[15];
  const float* n2g    = (const float*)d_in[16];
  const float* n2b    = (const float*)d_in[17];
  const float* fc1_w  = (const float*)d_in[18];
  const float* fc1_b  = (const float*)d_in[19];
  const float* fc2_w  = (const float*)d_in[20];
  const float* fc2_b  = (const float*)d_in[21];

  char* ws = (char*)d_ws;
  __bf16* wval  = (__bf16*)(ws + OFF_WVAL);
  __bf16* woa   = (__bf16*)(ws + OFF_WOA);
  __bf16* wout  = (__bf16*)(ws + OFF_WOUT);
  __bf16* wproj = (__bf16*)(ws + OFF_WPROJ);
  __bf16* wfc1  = (__bf16*)(ws + OFF_WFC1);
  __bf16* wfc2  = (__bf16*)(ws + OFF_WFC2);
  float*  boa   = (float*)(ws + OFF_BOA);
  float*  diag  = (float*)(ws + OFF_DIAG);
  float*  xn    = (float*)(ws + OFF_XN);     // xn (f32), later reused for z (proj out)
  float*  bufV  = (float*)(ws + OFF_V);      // value, later y
  __bf16* qb    = (__bf16*)(ws + OFF_Q);
  __bf16* bufBB = (__bf16*)(ws + OFF_BB);    // xn_bf16 -> attnpre -> h
  float*  bufOA = (float*)(ws + OFF_OA);     // off|aw (f32)
  __bf16* tbf   = (__bf16*)(ws + OFF_OA);    // reused as t (bf16, same bytes)
  __bf16* gbf   = (__bf16*)(ws + OFF_G);
  float*  outp  = (float*)d_out;

  // weights -> bf16, transposed to [N][K]
  pack_bf16_t<<<256, 256, 0, stream>>>(val_w,  wval,  256, 256);
  pack_bf16_t<<<256, 256, 0, stream>>>(out_w,  wout,  256, 256);
  pack_bf16_t<<<256, 256, 0, stream>>>(proj_w, wproj, 256, 256);
  pack_bf16_t<<<1024, 256, 0, stream>>>(fc1_w, wfc1, 256, 1024);
  pack_bf16_t<<<1024, 256, 0, stream>>>(fc2_w, wfc2, 1024, 256);
  pack_cat_t<<<128, 256, 0, stream>>>(off_w, aw_w, off_b, aw_b, woa, boa);

  // position embedding diagonal
  diag_kernel<<<196, 256, 0, stream>>>(mask, diag);

  // LN1 + window partition -> xn(f32), xn(bf16), q(bf16)
  ln1_kernel<<<Mtok / 8, 256, 0, stream>>>(x, n1g, n1b, diag, xn, bufBB, qb);

  // value = xn @ val_w + val_b   (f32 out, gathered later)
  gemm_wmma<0, 0, float><<<dim3(2, Mtok / 128), 256, 0, stream>>>(
      bufBB, wval, val_b, nullptr, bufV, Mtok, 256, 256);
  // [off|aw] = q @ [off_w|aw_w] + [off_b|aw_b]
  gemm_wmma<0, 0, float><<<dim3(1, Mtok / 128), 256, 0, stream>>>(
      qb, woa, boa, nullptr, bufOA, Mtok, 128, 256);

  // softmax + trilinear sampling -> attnpre (bf16)  [overwrites xn_bf16]
  sample_kernel<<<Mtok, 256, 0, stream>>>(bufV, bufOA, bufBB);

  // y = attnpre @ out_w + out_b + xn   (fused residual)  [overwrites value]
  gemm_wmma<0, 1, float><<<dim3(2, Mtok / 128), 256, 0, stream>>>(
      bufBB, wout, out_b, xn, bufV, Mtok, 256, 256);

  // t = LN(y)  (bf16)  [overwrites off|aw]
  ln_simple<<<Mtok / 8, 256, 0, stream>>>(bufV, ang, anb, tbf);

  // z = t @ proj_w + proj_b  (f32)  [overwrites xn]
  gemm_wmma<0, 0, float><<<dim3(2, Mtok / 128), 256, 0, stream>>>(
      tbf, wproj, proj_b, nullptr, xn, Mtok, 256, 256);

  // x_new = shortcut + reverse(z) -> d_out ; h = LN2(x_new) (bf16) [overwrites attnpre]
  reverse_ln2<<<Mtok / 8, 256, 0, stream>>>(x, xn, n2g, n2b, outp, bufBB);

  // g = gelu(h @ fc1_w + fc1_b)  (bf16)
  gemm_wmma<1, 0, __bf16><<<dim3(8, Mtok / 128), 256, 0, stream>>>(
      bufBB, wfc1, fc1_b, nullptr, gbf, Mtok, 1024, 256);

  // out = g @ fc2_w + fc2_b + x_new  (residual read from d_out, overwritten in place)
  gemm_wmma<0, 1, float><<<dim3(2, Mtok / 128), 256, 0, stream>>>(
      gbf, wfc2, fc2_b, outp, outp, Mtok, 256, 1024);
}
